// MIInfoNCESeqSeq_51264729645315
// MI455X (gfx1250) — matrix operand order repaired and checked by reading, not compiled, not run on hardware
//
#include <hip/hip_runtime.h>

typedef __attribute__((ext_vector_type(16))) _Float16 v16h;
typedef __attribute__((ext_vector_type(8)))  float    v8f;
typedef __attribute__((ext_vector_type(4)))  unsigned int u32x4;

union V16H { v16h v; u32x4 q[2]; };

#define Lpos 32
#define Bsz  256
#define H1n  128
#define H2n  64
#define ROWP 72    // pair-kernel LDS row stride (f16): 64 data + 8 pad
#define AP1  264   // mlp A row stride (f16): 256 + 8 pad
#define HP1  136   // mlp H row stride (f16): 128 + 8 pad

// CDNA5 16-bit A-matrix tile chunk (16x32 f16, one K-step):
// lane half 0 -> K base +0, half 1 -> +8; VGPR0-3 = K[0..7], VGPR4-7 = K[16..23].
__device__ __forceinline__ V16H ldA(const _Float16* rowbase, int hf) {
  V16H a;
  a.q[0] = *(const u32x4*)(rowbase + hf * 8);
  a.q[1] = *(const u32x4*)(rowbase + hf * 8 + 16);
  return a;
}
// CDNA5 16-bit B-matrix tile chunk (32x16, B[k][n] = row-major src row n):
// lane half 0 -> K base +0, half 1 -> +8; VGPR0-3 = K[0..7], VGPR4-7 = K[16..23].
__device__ __forceinline__ V16H ldB(const _Float16* rowbase, int hf) {
  V16H b;
  b.q[0] = *(const u32x4*)(rowbase + hf * 8);
  b.q[1] = *(const u32x4*)(rowbase + hf * 8 + 16);
  return b;
}
#define WMMA_F16(A, B, C) \
  __builtin_amdgcn_wmma_f32_16x16x32_f16(false, (A), false, (B), (short)0, (C), false, false)

// ---------------------------------------------------------------------------
// Prep 1: transpose inputs to position-major f16 rows. pos = i*B + b.
// Xt[sel][pos][c] = in[b, c, i]
// ---------------------------------------------------------------------------
__global__ __launch_bounds__(256)
void transpose_kernel(const float* __restrict__ x, const float* __restrict__ y,
                      _Float16* __restrict__ Xt) {
  const int sel = blockIdx.y;
  const int pos = blockIdx.x;            // i*256 + b
  const int b = pos & (Bsz - 1);
  const int i = pos >> 8;
  const int c = threadIdx.x;
  const float* inp = sel ? y : x;
  float v = inp[(size_t)b * (256 * Lpos) + (size_t)c * Lpos + i];
  Xt[(size_t)sel * 8192 * 256 + (size_t)pos * 256 + c] = (_Float16)v;
}

// ---------------------------------------------------------------------------
// Prep 2: convert weights to f16. Layout in Wc (f16 elems):
//   [0..32767]        W1g   [32768..65535]  W1h
//   [65536..73727]    W2g   [73728..81919]  W2h
// ---------------------------------------------------------------------------
__global__ __launch_bounds__(256)
void wcvt_kernel(const float* __restrict__ Wg1, const float* __restrict__ Wg2,
                 const float* __restrict__ Wh1, const float* __restrict__ Wh2,
                 _Float16* __restrict__ Wc) {
  const int idx = blockIdx.x * 256 + threadIdx.x;   // 128 blocks -> 32768
  Wc[idx]         = (_Float16)Wg1[idx];
  Wc[32768 + idx] = (_Float16)Wh1[idx];
  if (idx < 8192) {
    Wc[65536 + idx]        = (_Float16)Wg2[idx];
    Wc[65536 + 8192 + idx] = (_Float16)Wh2[idx];
  }
}

// ---------------------------------------------------------------------------
// MLP via WMMA. Block = 128 positions (8 waves x 1 p-tile of 16).
// Layer1: [16,256]x[256,128] +bias, ReLU -> LDS f16. Layer2: [16,128]x[128,64]
// +bias -> feat[pos][chan] f16. B-tiles read straight from L2-resident W.
// ---------------------------------------------------------------------------
__global__ __launch_bounds__(256)
void mlp_wmma_kernel(const _Float16* __restrict__ Xt, const _Float16* __restrict__ Wc,
                     const float* __restrict__ bg1, const float* __restrict__ bg2,
                     const float* __restrict__ bh1, const float* __restrict__ bh2,
                     _Float16* __restrict__ feat) {
  extern __shared__ char smem[];
  _Float16* As = (_Float16*)smem;            // [128][AP1]
  _Float16* Hs = As + 128 * AP1;             // [128][HP1]

  const int sel  = blockIdx.y;
  const int pos0 = blockIdx.x * 128;
  const int tid  = threadIdx.x;

  const _Float16* Asrc = Xt + (size_t)sel * 8192 * 256 + (size_t)pos0 * 256;
  const _Float16* W1h  = Wc + (size_t)sel * 32768;
  const _Float16* W2h  = Wc + 65536 + (size_t)sel * 8192;
  const float* b1p = sel ? bh1 : bg1;
  const float* b2p = sel ? bh2 : bg2;

  // stage A rows (128 x 256 f16, padded)
  {
    const int row = tid >> 1, hl = tid & 1;
    const u32x4* src = (const u32x4*)(Asrc + (size_t)row * 256 + hl * 128);
    u32x4* dst = (u32x4*)(As + row * AP1 + hl * 128);
    #pragma unroll
    for (int k = 0; k < 16; ++k) dst[k] = src[k];
  }
  __syncthreads();

  const int w = tid >> 5, t = tid & 31;
  const int mloc = t & 15, hf = t >> 4;
  const int prow0 = w * 16;

  // ---- layer 1: hoist all 8 A K-chunks into registers, reuse over 8 N-tiles
  V16H a1r[8];
  {
    const _Float16* arow = As + (prow0 + mloc) * AP1;
    #pragma unroll
    for (int ks = 0; ks < 8; ++ks) a1r[ks] = ldA(arow + ks * 32, hf);
  }
  #pragma unroll
  for (int nt = 0; nt < 8; ++nt) {
    v8f c = {};
    #pragma unroll
    for (int ks = 0; ks < 8; ++ks) {
      V16H b = ldB(W1h + (size_t)(nt * 16 + mloc) * 256 + ks * 32, hf);
      c = WMMA_F16(a1r[ks].v, b.v, c);
    }
    const float bv = b1p[nt * 16 + mloc];
    #pragma unroll
    for (int r = 0; r < 8; ++r) {
      float h = fmaxf(c[r] + bv, 0.0f);
      Hs[(prow0 + hf * 8 + r) * HP1 + nt * 16 + mloc] = (_Float16)h;
    }
  }
  // same-wave LDS ops are in-order; waves own disjoint H rows -> no barrier

  // ---- layer 2
  V16H a2r[4];
  {
    const _Float16* arow = Hs + (prow0 + mloc) * HP1;
    #pragma unroll
    for (int ks = 0; ks < 4; ++ks) a2r[ks] = ldA(arow + ks * 32, hf);
  }
  #pragma unroll
  for (int nt = 0; nt < 4; ++nt) {
    v8f c = {};
    #pragma unroll
    for (int ks = 0; ks < 4; ++ks) {
      V16H b = ldB(W2h + (size_t)(nt * 16 + mloc) * 128 + ks * 32, hf);
      c = WMMA_F16(a2r[ks].v, b.v, c);
    }
    const float bv = b2p[nt * 16 + mloc];
    #pragma unroll
    for (int r = 0; r < 8; ++r) {
      float o = c[r] + bv;
      const int pos = pos0 + prow0 + hf * 8 + r;
      feat[(size_t)sel * 524288 + (size_t)pos * 64 + nt * 16 + mloc] = (_Float16)o;
    }
  }
}

// ---------------------------------------------------------------------------
// Pair kernel: one block per (i,j). Two-pass fused max/diag then exp-sum,
// all cross-lane reductions hoisted out of the q-tile loop; B double-buffered.
// ---------------------------------------------------------------------------
__global__ __launch_bounds__(256)
void pair_kernel(const _Float16* __restrict__ xg, const _Float16* __restrict__ yh,
                 float* __restrict__ partials) {
  extern __shared__ char smem[];
  _Float16* Xs = (_Float16*)smem;          // [256][ROWP]
  _Float16* Ys = Xs + Bsz * ROWP;
  __shared__ float wred[8];

  const int blk = blockIdx.x;
  const int i = blk >> 5, j = blk & (Lpos - 1);
  const int tid = threadIdx.x;

  {
    const u32x4* gx = (const u32x4*)(xg + (size_t)i * Bsz * H2n + (size_t)tid * H2n);
    const u32x4* gy = (const u32x4*)(yh + (size_t)j * Bsz * H2n + (size_t)tid * H2n);
    u32x4* dx = (u32x4*)(Xs + tid * ROWP);
    u32x4* dy = (u32x4*)(Ys + tid * ROWP);
    #pragma unroll
    for (int k = 0; k < 8; ++k) { dx[k] = gx[k]; dy[k] = gy[k]; }
  }
  __syncthreads();

  const int w = tid >> 5, t = tid & 31;
  const int mloc = t & 15, hf = t >> 4;
  // C diag ownership: lanes 0..7 -> M=N=0..7 (vgpr r = lane);
  // lanes 24..31 -> M=N=8..15 (vgpr r = lane-24)
  const int rd = (t < 8) ? t : ((t >= 24) ? (t - 24) : -1);

  float acc = 0.0f;

  for (int iter = 0; iter < 2; ++iter) {
    const int pt = w + 8 * iter;

    V16H a0, a1;
    {
      const _Float16* yrow = Ys + (pt * 16 + mloc) * ROWP;
      a0 = ldA(yrow, hf);
      a1 = ldA(yrow + 32, hf);
    }

    // ---- pass 1: row max (per-lane running) + diagonal pick
    float mx[8];
    #pragma unroll
    for (int r = 0; r < 8; ++r) mx[r] = -__builtin_inff();
    {
      V16H b0, b1, nb0, nb1;
      const _Float16* xr0 = Xs + mloc * ROWP;
      b0 = ldB(xr0, hf); b1 = ldB(xr0 + 32, hf);
      for (int qt = 0; qt < 16; ++qt) {
        if (qt < 15) {
          const _Float16* xr = Xs + ((qt + 1) * 16 + mloc) * ROWP;
          nb0 = ldB(xr, hf); nb1 = ldB(xr + 32, hf);
        }
        v8f c = {};
        c = WMMA_F16(a0.v, b0.v, c);
        c = WMMA_F16(a1.v, b1.v, c);
        if (qt == pt) {
          #pragma unroll
          for (int r = 0; r < 8; ++r) acc += (r == rd) ? c[r] : 0.0f;
        }
        #pragma unroll
        for (int r = 0; r < 8; ++r) mx[r] = fmaxf(mx[r], c[r]);
        b0 = nb0; b1 = nb1;
      }
    }
    // one reduction per p-tile: broadcast row max to all 16 lanes of each half
    #pragma unroll
    for (int r = 0; r < 8; ++r) {
      #pragma unroll
      for (int o = 8; o >= 1; o >>= 1) mx[r] = fmaxf(mx[r], __shfl_xor(mx[r], o, 16));
    }

    // ---- pass 2: exp-sum with known max (pure per-lane accumulation)
    float sm[8];
    #pragma unroll
    for (int r = 0; r < 8; ++r) sm[r] = 0.0f;
    {
      V16H b0, b1, nb0, nb1;
      const _Float16* xr0 = Xs + mloc * ROWP;
      b0 = ldB(xr0, hf); b1 = ldB(xr0 + 32, hf);
      for (int qt = 0; qt < 16; ++qt) {
        if (qt < 15) {
          const _Float16* xr = Xs + ((qt + 1) * 16 + mloc) * ROWP;
          nb0 = ldB(xr, hf); nb1 = ldB(xr + 32, hf);
        }
        v8f c = {};
        c = WMMA_F16(a0.v, b0.v, c);
        c = WMMA_F16(a1.v, b1.v, c);
        #pragma unroll
        for (int r = 0; r < 8; ++r) sm[r] += __expf(c[r] - mx[r]);
        b0 = nb0; b1 = nb1;
      }
    }
    #pragma unroll
    for (int r = 0; r < 8; ++r) {
      #pragma unroll
      for (int o = 8; o >= 1; o >>= 1) sm[r] += __shfl_xor(sm[r], o, 16);
      if (mloc == 0) acc -= mx[r] + __logf(sm[r]);   // lanes 0 & 16: rows M=r, 8+r
    }
  }

  #pragma unroll
  for (int o = 16; o >= 1; o >>= 1) acc += __shfl_xor(acc, o, 32);
  if (t == 0) wred[w] = acc;
  __syncthreads();
  if (tid == 0) {
    float s = 0.0f;
    #pragma unroll
    for (int k = 0; k < 8; ++k) s += wred[k];
    partials[blk] = s;
  }
}

// ---------------------------------------------------------------------------
// Deterministic fixed-order reduction of 1024 partials -> scalar.
// ---------------------------------------------------------------------------
__global__ __launch_bounds__(256)
void finalize_kernel(const float* __restrict__ partials, float* __restrict__ out) {
  __shared__ float wred[8];
  const int t = threadIdx.x;
  float s = partials[t] + partials[t + 256] + partials[t + 512] + partials[t + 768];
  #pragma unroll
  for (int o = 16; o >= 1; o >>= 1) s += __shfl_xor(s, o, 32);
  if ((t & 31) == 0) wred[t >> 5] = s;
  __syncthreads();
  if (t == 0) {
    float tot = 0.0f;
    #pragma unroll
    for (int k = 0; k < 8; ++k) tot += wred[k];
    out[0] = __logf(256.0f) + tot / (1024.0f * 256.0f);
  }
}

extern "C" void kernel_launch(void* const* d_in, const int* in_sizes, int n_in,
                              void* d_out, int out_size, void* d_ws, size_t ws_size,
                              hipStream_t stream) {
  (void)in_sizes; (void)n_in; (void)out_size; (void)ws_size;
  const float* x   = (const float*)d_in[0];
  const float* y   = (const float*)d_in[1];
  const float* Wg1 = (const float*)d_in[2];
  const float* bg1 = (const float*)d_in[3];
  const float* Wg2 = (const float*)d_in[4];
  const float* bg2 = (const float*)d_in[5];
  const float* Wh1 = (const float*)d_in[6];
  const float* bh1 = (const float*)d_in[7];
  const float* Wh2 = (const float*)d_in[8];
  const float* bh2 = (const float*)d_in[9];

  // workspace carve (bytes)
  char* ws = (char*)d_ws;
  _Float16* feat = (_Float16*)ws;                          // 2*524288 f16 = 2 MB
  _Float16* Xt   = (_Float16*)(ws + 2097152);              // 2*8192*256 f16 = 8 MB
  _Float16* Wc   = (_Float16*)(ws + 2097152 + 8388608);    // 81920 f16
  float* partials = (float*)(ws + 2097152 + 8388608 + 163840); // 4 KB

  transpose_kernel<<<dim3(8192, 2), 256, 0, stream>>>(x, y, Xt);
  wcvt_kernel<<<128, 256, 0, stream>>>(Wg1, Wg2, Wh1, Wh2, Wc);

  const size_t sh_mlp = (size_t)(128 * AP1 + 128 * HP1) * sizeof(_Float16); // 102400 B
  mlp_wmma_kernel<<<dim3(64, 2), 256, sh_mlp, stream>>>(Xt, Wc, bg1, bg2, bh1, bh2, feat);

  const _Float16* xg = feat;
  const _Float16* yh = feat + 524288;
  const size_t sh_pair = (size_t)2 * Bsz * ROWP * sizeof(_Float16);         // 73728 B
  pair_kernel<<<Lpos * Lpos, 256, sh_pair, stream>>>(xg, yh, partials);

  finalize_kernel<<<1, 256, 0, stream>>>(partials, (float*)d_out);
}
